// MoE_45475113730386
// MI455X (gfx1250) — matrix operand order repaired and checked by reading, not compiled, not run on hardware
//
#include <hip/hip_runtime.h>

// ---------------------------------------------------------------------------
// Types (raw ext_vector types: trivially-copyable, union-safe)
// ---------------------------------------------------------------------------
typedef __attribute__((ext_vector_type(16))) __bf16        v16bf;
typedef __attribute__((ext_vector_type(8)))  float         v8f;
typedef __attribute__((ext_vector_type(4)))  unsigned int  u32x4;
typedef __attribute__((ext_vector_type(4)))  float         f32x4;
typedef __attribute__((ext_vector_type(4)))  int           v4i;

union FragBF16 {           // 32 bytes: 16 bf16 elements, loadable as 2x b128
    u32x4 u[2];
    v16bf v;
};

#define N_TOK   8192
#define DMODEL  1024
#define NEXP    8

// ---------------------------------------------------------------------------
// CDNA5 async global->LDS path (ASYNCcnt), with fallback to sync copy
// Builtin signature (from toolchain diagnostic): (v4i AS1*, v4i AS3*, Ii, Ii)
// ---------------------------------------------------------------------------
#if __has_builtin(__builtin_amdgcn_global_load_async_to_lds_b128)
#define USE_ASYNC_LDS 1
typedef __attribute__((address_space(1))) v4i as1_v4i;
typedef __attribute__((address_space(3))) v4i as3_v4i;
#define ASYNC_B128(gsrc, ldst)                                                 \
    __builtin_amdgcn_global_load_async_to_lds_b128((as1_v4i*)(gsrc),           \
                                                   (as3_v4i*)(ldst), 0, 0)
#if __has_builtin(__builtin_amdgcn_s_wait_asynccnt)
#define WAIT_ASYNC(n) __builtin_amdgcn_s_wait_asynccnt(n)
#else
#define WAIT_ASYNC(n) asm volatile("s_wait_asynccnt %0" :: "n"(n) : "memory")
#endif
#else
#define USE_ASYNC_LDS 0
#define WAIT_ASYNC(n)
#endif

__device__ __forceinline__ unsigned short f32_to_bf16_rne(float f) {
    unsigned int u = __float_as_uint(f);
    u = (u + 0x7FFFu + ((u >> 16) & 1u)) >> 16;
    return (unsigned short)u;
}

// ---------------------------------------------------------------------------
// f32 -> bf16 bulk convert (x activations)
// ---------------------------------------------------------------------------
__global__ __launch_bounds__(256)
void convert_x_kernel(const float* __restrict__ x, unsigned short* __restrict__ xb) {
    size_t i = ((size_t)blockIdx.x * 256 + threadIdx.x) * 4;
    f32x4 v = *(const f32x4*)(x + i);
    unsigned short o0 = f32_to_bf16_rne(v.x);
    unsigned short o1 = f32_to_bf16_rne(v.y);
    unsigned short o2 = f32_to_bf16_rne(v.z);
    unsigned short o3 = f32_to_bf16_rne(v.w);
    unsigned long long packed =
        (unsigned long long)o0 | ((unsigned long long)o1 << 16) |
        ((unsigned long long)o2 << 32) | ((unsigned long long)o3 << 48);
    *(unsigned long long*)(xb + i) = packed;
}

// ---------------------------------------------------------------------------
// Weight convert + transpose: W[k][n] f32 (row-major [in,out]) -> Wt[n][k] bf16
// ---------------------------------------------------------------------------
__global__ __launch_bounds__(256)
void convert_transpose_w_kernel(const float* __restrict__ W,
                                unsigned short* __restrict__ Wt) {
    __shared__ float tile[32][33];
    int tx = threadIdx.x & 31;
    int ty = threadIdx.x >> 5;              // 0..7
    int k0 = blockIdx.x * 32;
    int n0 = blockIdx.y * 32;
#pragma unroll
    for (int i = 0; i < 4; ++i) {
        int k = ty + i * 8;
        tile[k][tx] = W[(size_t)(k0 + k) * DMODEL + n0 + tx];
    }
    __syncthreads();
#pragma unroll
    for (int i = 0; i < 4; ++i) {
        int n = ty + i * 8;
        Wt[(size_t)(n0 + n) * DMODEL + k0 + tx] = f32_to_bf16_rne(tile[tx][n]);
    }
}

// ---------------------------------------------------------------------------
// Gating: logits = x@Wg + ea@We + b ; top-2 ; softmax over the 2
// ---------------------------------------------------------------------------
__global__ __launch_bounds__(256)
void gating_kernel(const float* __restrict__ x, const float* __restrict__ ea,
                   const float* __restrict__ wg, const float* __restrict__ ew,
                   const float* __restrict__ eb,
                   float* __restrict__ gates, int* __restrict__ topi) {
    int wave = threadIdx.x >> 5;
    int lane = threadIdx.x & 31;
    int n = blockIdx.x * 8 + wave;
    const float* xr = x + (size_t)n * DMODEL;

    float acc[NEXP] = {0.f, 0.f, 0.f, 0.f, 0.f, 0.f, 0.f, 0.f};
    for (int k = lane; k < DMODEL; k += 32) {
        float xv = xr[k];
        const float* wr = wg + (size_t)k * NEXP;
#pragma unroll
        for (int e = 0; e < NEXP; ++e) acc[e] += xv * wr[e];
    }
#pragma unroll
    for (int off = 16; off > 0; off >>= 1) {
#pragma unroll
        for (int e = 0; e < NEXP; ++e)
            acc[e] += __shfl_xor(acc[e], off, 32);
    }
    if (lane == 0) {
        float ea0 = ea[n * 2], ea1 = ea[n * 2 + 1];
        float lg[NEXP];
#pragma unroll
        for (int e = 0; e < NEXP; ++e)
            lg[e] = acc[e] + ea0 * ew[e] + ea1 * ew[NEXP + e] + eb[e];
        float b0 = -3.4e38f; int i0 = 0;
#pragma unroll
        for (int e = 0; e < NEXP; ++e) if (lg[e] > b0) { b0 = lg[e]; i0 = e; }
        float b1 = -3.4e38f; int i1 = 0;
#pragma unroll
        for (int e = 0; e < NEXP; ++e) if (e != i0 && lg[e] > b1) { b1 = lg[e]; i1 = e; }
        float e1 = __expf(b1 - b0);          // <= 1
        float inv = 1.0f / (1.0f + e1);
        gates[n * 2]     = inv;
        gates[n * 2 + 1] = e1 * inv;
        topi[n * 2]      = i0;
        topi[n * 2 + 1]  = i1;
    }
}

// ---------------------------------------------------------------------------
// WMMA GEMM: C[8192 x 1024] = A_bf16 @ Wt_bf16^T + bias
//   Block tile 256(M) x 128(N); 8 waves, each wave 32 rows x 128 cols:
//   2 A fragments, 16 accumulators, each B fragment feeds 2 WMMAs.
//   B slab (128 cols x 32 k) double-buffered in LDS via async DMA loads.
// ---------------------------------------------------------------------------
#define LDSB_STRIDE 40   // bf16 per row; 80B rows -> conflict-free ds_load_b128

__global__ __launch_bounds__(256)
void gemm_bf16_wmma(const unsigned short* __restrict__ A,
                    const unsigned short* __restrict__ Bt,
                    const float* __restrict__ bias,
                    float* __restrict__ Cf,             // may be null
                    unsigned short* __restrict__ Crelu) // may be null
{
    __shared__ unsigned short ldsB[2][128 * LDSB_STRIDE];

    const int tid  = threadIdx.x;
    const int wave = tid >> 5;
    const int lane = tid & 31;
    const int mBlock = blockIdx.x * 256;
    const int nBlock = blockIdx.y * 128;

    const int colLane = lane & 15;
    const int khalf   = lane >> 4;                 // A/B K-half select

    const int mRow0 = mBlock + wave * 32 + colLane; // A frag rows: lane%16 = M
    v8f acc[2][8] = {};

    const u32x4* Arow0 = (const u32x4*)(A + (size_t)mRow0 * DMODEL);
    const u32x4* Arow1 = (const u32x4*)(A + (size_t)(mRow0 + 16) * DMODEL);

    // cooperative staging: thread t covers col (t>>1), k-half (t&1) of slab
    const int nCoop    = tid >> 1;
    const int halfCoop = tid & 1;
    const unsigned short* bSrcBase =
        Bt + (size_t)(nBlock + nCoop) * DMODEL + halfCoop * 16;
    unsigned short* bDst0 = &ldsB[0][nCoop * LDSB_STRIDE + halfCoop * 16];
    unsigned short* bDst1 = &ldsB[1][nCoop * LDSB_STRIDE + halfCoop * 16];

    // prologue: stage slab 0 into buffer 0
#if USE_ASYNC_LDS
    ASYNC_B128(bSrcBase, bDst0);
    ASYNC_B128(bSrcBase + 8, bDst0 + 8);
#else
    {
        const u32x4* s = (const u32x4*)bSrcBase;
        u32x4* d = (u32x4*)bDst0;
        d[0] = s[0]; d[1] = s[1];
    }
#endif

    int buf = 0;
    for (int k0 = 0; k0 < DMODEL; k0 += 32, buf ^= 1) {
        // stage next slab into the other buffer (overlaps with this compute)
        if (k0 + 32 < DMODEL) {
            const unsigned short* s = bSrcBase + (k0 + 32);
            unsigned short* d = buf ? bDst0 : bDst1;
#if USE_ASYNC_LDS
            ASYNC_B128(s, d);
            ASYNC_B128(s + 8, d + 8);
            WAIT_ASYNC(2);   // oldest 2 (current slab) retired; next 2 in flight
#else
            const u32x4* sv = (const u32x4*)s;
            u32x4* dv = (u32x4*)d;
            dv[0] = sv[0]; dv[1] = sv[1];
#endif
        } else {
            WAIT_ASYNC(0);
        }
        __syncthreads();     // all waves' slab writes visible

        // A fragments: 16x32 bf16 each (ISA 7.12.2 16-bit A layout)
        FragBF16 a0, a1;
        a0.u[0] = Arow0[(k0 + khalf * 8) >> 3];
        a0.u[1] = Arow0[(k0 + khalf * 8 + 16) >> 3];
        a1.u[0] = Arow1[(k0 + khalf * 8) >> 3];
        a1.u[1] = Arow1[(k0 + khalf * 8 + 16) >> 3];

#pragma unroll
        for (int nt = 0; nt < 8; ++nt) {
            FragBF16 b;   // 32x16 bf16: lane = col, K = 8h..8h+7 and 16+8h..16+8h+7
            const u32x4* bp = (const u32x4*)(&ldsB[buf][(nt * 16 + colLane) * LDSB_STRIDE
                                                        + khalf * 8]);
            b.u[0] = bp[0];
            b.u[1] = bp[2];
            acc[0][nt] = __builtin_amdgcn_wmma_f32_16x16x32_bf16(
                false, a0.v, false, b.v, (short)0, acc[0][nt], false, false);
            acc[1][nt] = __builtin_amdgcn_wmma_f32_16x16x32_bf16(
                false, a1.v, false, b.v, (short)0, acc[1][nt], false, false);
        }
        __syncthreads();     // protect LDS buffer against next-iter overwrite
    }

    // Epilogue: C/D layout — col = lane%16, row = VGPR r + 8*(lane/16)
    const int rowBase = mBlock + wave * 32 + (lane >> 4) * 8;
    if (Cf) {
#pragma unroll
        for (int mh = 0; mh < 2; ++mh)
#pragma unroll
            for (int nt = 0; nt < 8; ++nt) {
                int col = nBlock + nt * 16 + colLane;
                float bv = bias[col];
#pragma unroll
                for (int r = 0; r < 8; ++r)
                    Cf[(size_t)(rowBase + mh * 16 + r) * DMODEL + col] =
                        acc[mh][nt][r] + bv;
            }
    }
    if (Crelu) {
#pragma unroll
        for (int mh = 0; mh < 2; ++mh)
#pragma unroll
            for (int nt = 0; nt < 8; ++nt) {
                int col = nBlock + nt * 16 + colLane;
                float bv = bias[col];
#pragma unroll
                for (int r = 0; r < 8; ++r) {
                    float v = acc[mh][nt][r] + bv;
                    Crelu[(size_t)(rowBase + mh * 16 + r) * DMODEL + col] =
                        f32_to_bf16_rne(v > 0.f ? v : 0.f);
                }
            }
    }
}

// ---------------------------------------------------------------------------
// Combine: final[n][:] = g0 * expert_out[i0][n][:] + g1 * expert_out[i1][n][:]
// ---------------------------------------------------------------------------
__global__ __launch_bounds__(256)
void combine_kernel(const float* __restrict__ expert_out,
                    const float* __restrict__ gates,
                    const int* __restrict__ topi,
                    float* __restrict__ out) {
    size_t gid  = (size_t)blockIdx.x * 256 + threadIdx.x;  // over N*D/4
    size_t base = gid * 4;
    int n = (int)(base / DMODEL);
    int i0 = topi[n * 2], i1 = topi[n * 2 + 1];
    float g0 = gates[n * 2], g1 = gates[n * 2 + 1];
    const size_t ESTRIDE = (size_t)N_TOK * DMODEL;
    f32x4 a = *(const f32x4*)(expert_out + (size_t)i0 * ESTRIDE + base);
    f32x4 b = *(const f32x4*)(expert_out + (size_t)i1 * ESTRIDE + base);
    f32x4 r;
    r.x = g0 * a.x + g1 * b.x;
    r.y = g0 * a.y + g1 * b.y;
    r.z = g0 * a.z + g1 * b.z;
    r.w = g0 * a.w + g1 * b.w;
    *(f32x4*)(out + base) = r;
}

// ---------------------------------------------------------------------------
// Loss: importance/load histograms + cv^2, deterministic single-block reduce
// ---------------------------------------------------------------------------
__device__ __forceinline__ float cv2_8(const float* v) {
    float s = 0.f;
#pragma unroll
    for (int e = 0; e < NEXP; ++e) s += v[e];
    float m = s * (1.0f / NEXP);
    float var = 0.f;
#pragma unroll
    for (int e = 0; e < NEXP; ++e) { float d = v[e] - m; var += d * d; }
    var *= (1.0f / (NEXP - 1));              // unbiased (ddof=1)
    return var / (m * m + 1e-10f);
}

__global__ __launch_bounds__(256)
void loss_kernel(const float* __restrict__ gates, const int* __restrict__ topi,
                 float* __restrict__ out_loss) {
    __shared__ float simp[256][NEXP];
    __shared__ float sld[256][NEXP];
    int tid = threadIdx.x;
    float imp[NEXP] = {0.f, 0.f, 0.f, 0.f, 0.f, 0.f, 0.f, 0.f};
    float ld[NEXP]  = {0.f, 0.f, 0.f, 0.f, 0.f, 0.f, 0.f, 0.f};
    for (int n = tid; n < N_TOK; n += 256) {
#pragma unroll
        for (int k = 0; k < 2; ++k) {
            int e = topi[n * 2 + k];
            imp[e] += gates[n * 2 + k];
            ld[e]  += 1.0f;
        }
    }
#pragma unroll
    for (int e = 0; e < NEXP; ++e) { simp[tid][e] = imp[e]; sld[tid][e] = ld[e]; }
    __syncthreads();
    for (int s = 128; s > 0; s >>= 1) {
        if (tid < s) {
#pragma unroll
            for (int e = 0; e < NEXP; ++e) {
                simp[tid][e] += simp[tid + s][e];
                sld[tid][e]  += sld[tid + s][e];
            }
        }
        __syncthreads();
    }
    if (tid == 0)
        *out_loss = 0.01f * (cv2_8(simp[0]) + cv2_8(sld[0]));
}

// ---------------------------------------------------------------------------
// Host launcher
// d_in order (setup_inputs dict, pytree-flattened):
//   [0]=x [1]=edge_attr [2]=w_gating [3]=edge_w [4]=edge_b
//   [5..34] = 15 x (w, b) pairs for experts with layer counts {1,2,3,1,2,3,1,2}
// d_out: final[8192*1024] | expert_out[8*8192*1024] | loss[1]  (f32)
// ---------------------------------------------------------------------------
extern "C" void kernel_launch(void* const* d_in, const int* in_sizes, int n_in,
                              void* d_out, int out_size, void* d_ws, size_t ws_size,
                              hipStream_t stream) {
    (void)in_sizes; (void)n_in; (void)out_size; (void)ws_size;

    const float* x      = (const float*)d_in[0];
    const float* ea     = (const float*)d_in[1];
    const float* wg     = (const float*)d_in[2];
    const float* ew     = (const float*)d_in[3];
    const float* eb     = (const float*)d_in[4];

    static const int nLayers[NEXP] = {1, 2, 3, 1, 2, 3, 1, 2};  // eid%3 cycle
    const size_t MB = 1024ULL * 1024ULL;

    // workspace layout
    char* ws = (char*)d_ws;
    unsigned short* xb    = (unsigned short*)(ws);                 // 16 MB bf16 x
    unsigned short* wt    = (unsigned short*)(ws + 16 * MB);       // 30 MB bf16 Wt x15
    unsigned short* h0    = (unsigned short*)(ws + 46 * MB);       // 16 MB hidden A
    unsigned short* h1    = (unsigned short*)(ws + 62 * MB);       // 16 MB hidden B
    float*          gates = (float*)(ws + 78 * MB);                // 64 KB
    int*            topi  = (int*)(ws + 78 * MB + (size_t)N_TOK * 2 * sizeof(float));

    float* final_out  = (float*)d_out;
    float* expert_out = final_out + (size_t)N_TOK * DMODEL;
    float* loss_out   = expert_out + (size_t)NEXP * N_TOK * DMODEL;

    // 1) convert x -> bf16
    convert_x_kernel<<<(N_TOK * DMODEL) / (256 * 4), 256, 0, stream>>>(x, xb);

    // 2) convert+transpose all 15 weight matrices -> bf16 Wt[n][k]
    {
        int li = 0;
        for (int e = 0; e < NEXP; ++e)
            for (int j = 0; j < nLayers[e]; ++j, ++li)
                convert_transpose_w_kernel<<<dim3(32, 32), 256, 0, stream>>>(
                    (const float*)d_in[5 + 2 * li], wt + (size_t)li * DMODEL * DMODEL);
    }

    // 3) gating (top-2 + softmax)
    gating_kernel<<<N_TOK / 8, 256, 0, stream>>>(x, ea, wg, ew, eb, gates, topi);

    // 4) expert GEMM chains (WMMA bf16 -> f32)
    {
        int li = 0;
        dim3 grid(N_TOK / 256, DMODEL / 128);
        for (int e = 0; e < NEXP; ++e) {
            const unsigned short* Ain = xb;
            for (int j = 0; j < nLayers[e]; ++j, ++li) {
                const float* bias = (const float*)d_in[6 + 2 * li];
                bool last = (j == nLayers[e] - 1);
                float* Cf = last ? (expert_out + (size_t)e * N_TOK * DMODEL) : nullptr;
                unsigned short* Cr = last ? nullptr : ((j & 1) ? h1 : h0);
                gemm_bf16_wmma<<<grid, 256, 0, stream>>>(
                    Ain, wt + (size_t)li * DMODEL * DMODEL, bias, Cf, Cr);
                if (!last) Ain = Cr;
            }
        }
    }

    // 5) combine top-2 expert outputs
    combine_kernel<<<(N_TOK * DMODEL / 4) / 256, 256, 0, stream>>>(
        expert_out, gates, topi, final_out);

    // 6) aux loss
    loss_kernel<<<1, 256, 0, stream>>>(gates, topi, loss_out);
}
